// RNNet_52269751992850
// MI455X (gfx1250) — matrix operand order repaired and checked by reading, not compiled, not run on hardware
//
#include <hip/hip_runtime.h>
#include <hip/hip_bf16.h>
#include <cstdint>
#include <cstddef>

// ---------------------------------------------------------------------------
// MI455X (gfx1250) LSTM inference:
//   phase0 (prep):  W_ih,W_hh -> bf16; bias = b_ih+b_hh; head collapsed to
//                   v = W2@W1 (128), s = W2.b1 + b2 (exact: no nonlinearity
//                   between the dense layers in the reference).
//   phase1 (GEMM):  xproj[t][b][512] (bf16, in ws, L2-resident ~134MB)
//                   = transpose(x) @ W_ih^T + bias   via v_wmma_f32_16x16x32_bf16
//   phase2 (scan):  persistent blocks, W_hh in LDS, c in VGPRs, h via LDS,
//                   per-step xproj tile fetched by the Tensor Data Mover
//                   (tensor_load_to_lds + s_wait_tensorcnt), WMMA recurrence.
// ---------------------------------------------------------------------------

typedef __attribute__((ext_vector_type(16))) __bf16         v16bf;
typedef __attribute__((ext_vector_type(8)))  float          v8f;
typedef __attribute__((ext_vector_type(8)))  unsigned short ush8;
typedef __attribute__((ext_vector_type(4)))  unsigned short ush4;
typedef __attribute__((ext_vector_type(4)))  unsigned int   u32x4;
typedef __attribute__((ext_vector_type(8)))  int            i32x8;
typedef __attribute__((ext_vector_type(4)))  int            i32x4;

union FragU { ush8 h[2]; v16bf v; };

__device__ __forceinline__ unsigned short f2bf(float f) {
  unsigned int u = __builtin_bit_cast(unsigned int, f);
  u += 0x7FFFu + ((u >> 16) & 1u);   // round-to-nearest-even
  return (unsigned short)(u >> 16);
}
__device__ __forceinline__ float bf2f(unsigned short u) {
  unsigned int x = ((unsigned int)u) << 16;
  return __builtin_bit_cast(float, x);
}
__device__ __forceinline__ float sigmoidf_(float x) {
  return 1.0f / (1.0f + __expf(-x));
}
__device__ __forceinline__ v8f wmma_bf16(v16bf a, v16bf b, v8f c) {
  // D = A(16x32) * B(32x16) + C, f32 accumulate
  return __builtin_amdgcn_wmma_f32_16x16x32_bf16(false, a, false, b, (short)0, c, false, false);
}

// A fragment (16x32 bf16, M x K) from row-major half matrix [rows][pitch].
// ISA layout: lanes 0-15 -> M=lane, halves 0-7 = K k0..k0+7, halves 8-15 =
// K k0+16..k0+23; lanes 16-31 shift K base by +8.  => two b128 loads per lane.
__device__ __forceinline__ v16bf ldA(const unsigned short* p, int pitch,
                                     int row0, int k0, int lane) {
  int r  = row0 + (lane & 15);
  int kb = k0 + ((lane >> 4) << 3);
  FragU f;
  f.h[0] = *(const ush8*)(p + r * pitch + kb);
  f.h[1] = *(const ush8*)(p + r * pitch + kb + 16);
  return f.v;
}

// B fragment (32x16 bf16, K x N) where source is row-major W[N][K]
// (B[k][n] = W[n][k]).  Lane = column n, halves 0-7 = K kb..kb+7, 8-15 = K kb+16..kb+23, kb = k0 + 8*(lane/16).
__device__ __forceinline__ v16bf ldB(const unsigned short* p, int pitch,
                                     int n0, int k0, int lane) {
  int n  = n0 + (lane & 15);
  int kb = k0 + ((lane >> 4) << 3);
  const ush8* q = (const ush8*)(p + n * pitch + kb);
  FragU f;
  f.h[0] = q[0];
  f.h[1] = q[2];
  return f.v;
}

static constexpr int B_   = 4096;
static constexpr int T_   = 32;
static constexpr int D_   = 288;   // 9 * 32
static constexpr int H_   = 128;
static constexpr int G4_  = 512;   // 4*H
static constexpr int XPCH = 296;   // phase1 LDS pitch (halves), 288 + 8 pad
static constexpr int KP2  = 136;   // phase2 W_hh / h pitch (halves), 128 + 8
static constexpr int XPP2 = 512;   // phase2 xproj tile pitch (halves) — exact
                                   // row width so the TDM copy is contiguous

// ---------------------------------------------------------------------------
// phase 0: weight conversion + head collapse
// ---------------------------------------------------------------------------
__global__ __launch_bounds__(256)
void lstm_prep_kernel(const float* __restrict__ Wih, const float* __restrict__ Whh,
                      const float* __restrict__ bih, const float* __restrict__ bhh,
                      const float* __restrict__ W1,  const float* __restrict__ b1,
                      const float* __restrict__ W2,  const float* __restrict__ b2,
                      unsigned short* __restrict__ wih_bf,
                      unsigned short* __restrict__ whh_bf,
                      float* __restrict__ bias, float* __restrict__ vh,
                      float* __restrict__ sbias) {
  const int tid    = blockIdx.x * blockDim.x + threadIdx.x;
  const int stride = gridDim.x * blockDim.x;
  for (int q = tid; q < G4_ * D_; q += stride) wih_bf[q] = f2bf(Wih[q]);
  for (int q = tid; q < G4_ * H_; q += stride) whh_bf[q] = f2bf(Whh[q]);
  for (int q = tid; q < G4_;      q += stride) bias[q]   = bih[q] + bhh[q];
  if (blockIdx.x == 0) {
    for (int q = threadIdx.x; q < H_; q += blockDim.x) {
      float acc = 0.f;
      for (int j = 0; j < 256; ++j) acc += W2[j] * W1[j * H_ + q];
      vh[q] = acc;
    }
    if (threadIdx.x == 0) {
      float acc = b2[0];
      for (int j = 0; j < 256; ++j) acc += W2[j] * b1[j];
      sbias[0] = acc;
    }
  }
}

// ---------------------------------------------------------------------------
// phase 1: xproj[t][b][g] = (x transposed) @ W_ih^T + bias, bf16 out
//   grid: (B*T)/128 = 1024 blocks, 256 threads (8 waves, 16 M-rows each)
// ---------------------------------------------------------------------------
__global__ __launch_bounds__(256)
void lstm_xproj_kernel(const float* __restrict__ x,
                       const unsigned short* __restrict__ wih_bf,
                       const float* __restrict__ bias,
                       unsigned short* __restrict__ xp) {
  extern __shared__ char smem[];
  unsigned short* xA = (unsigned short*)smem;      // [128][XPCH] bf16 A tile
  unsigned short* Wb = xA + 128 * XPCH;            // [128][XPCH] bf16 B chunk

  const int tid  = threadIdx.x;
  const int lane = tid & 31;
  const int wave = tid >> 5;
  const int m0   = blockIdx.x * 128;               // m = b*T + t

  // Stage A: gather x[b][c][t][nb] -> row-major [m_local][d] bf16 in LDS.
  for (int q = tid; q < 128 * 72; q += 256) {      // 72 float4 per row (288/4)
    int r = q / 72, k = (q % 72) * 4;
    int m = m0 + r;
    int b = m >> 5, t = m & 31;
    int c = k >> 5, nb = k & 31;
    const float4 v4 = *(const float4*)(x + ((((size_t)b * 9 + c) * 32 + t) << 5) + nb);
    ush4 h;
    h[0] = f2bf(v4.x); h[1] = f2bf(v4.y); h[2] = f2bf(v4.z); h[3] = f2bf(v4.w);
    *(ush4*)(xA + r * XPCH + k) = h;
  }
  __syncthreads();

  // A fragments resident in VGPRs for the whole block (9 x v16bf = 72 VGPRs).
  v16bf afr[9];
#pragma unroll
  for (int kb = 0; kb < 9; ++kb) afr[kb] = ldA(xA, XPCH, wave * 16, kb * 32, lane);

  for (int nc = 0; nc < 4; ++nc) {                 // 4 x 128 output columns
    __syncthreads();                               // previous chunk consumed
    for (int q = tid; q < 128 * 36; q += 256) {    // 36 ush8 per W row (288/8)
      int r = q / 36, k8 = (q % 36) * 8;
      *(ush8*)(Wb + r * XPCH + k8) =
          *(const ush8*)(wih_bf + ((size_t)(nc * 128 + r)) * D_ + k8);
    }
    __syncthreads();

#pragma unroll
    for (int nt = 0; nt < 8; ++nt) {
      v8f acc = {0.f, 0.f, 0.f, 0.f, 0.f, 0.f, 0.f, 0.f};
#pragma unroll
      for (int kb = 0; kb < 9; ++kb)
        acc = wmma_bf16(afr[kb], ldB(Wb, XPCH, nt * 16, kb * 32, lane), acc);

      const int col  = nc * 128 + nt * 16 + (lane & 15);
      const float bv = bias[col];
      const int mloc = wave * 16 + ((lane >> 4) << 3);
#pragma unroll
      for (int v = 0; v < 8; ++v) {                // D layout: M = v (+8), N = lane%16
        int m = m0 + mloc + v;
        int b = m >> 5, t = m & 31;
        xp[((size_t)t * B_ + b) * G4_ + col] = f2bf(acc[v] + bv);
      }
    }
  }
}

// ---------------------------------------------------------------------------
// phase 2: persistent LSTM scan.  grid: 128 blocks x 256 threads.
//   block owns 32 batch rows; wave w: M-tile = (w>>2)*16, nt-pair = (w&3)*2.
//   LDS: W_hh [512][KP2] (139KB, loaded once) + h [32][KP2] + xp tile [32][512]
//   The per-step xp tile (32KB, contiguous) is moved by the Tensor Data Mover.
// ---------------------------------------------------------------------------
__global__ __launch_bounds__(256)
void lstm_scan_kernel(const unsigned short* __restrict__ xp,
                      const unsigned short* __restrict__ whh_bf,
                      const float* __restrict__ vh, const float* __restrict__ sbias,
                      float* __restrict__ out) {
  extern __shared__ char smem[];
  unsigned short* Whh  = (unsigned short*)smem;    // [512][KP2]
  unsigned short* hbuf = Whh + 512 * KP2;          // [32][KP2]
  unsigned short* xpb  = hbuf + 32 * KP2;          // [32][XPP2]
  constexpr unsigned XPB_LDS_OFF = (unsigned)((512 * KP2 + 32 * KP2) * 2); // bytes

  const int tid    = threadIdx.x;
  const int lane   = tid & 31;
  const int wave   = tid >> 5;
  const int b0     = blockIdx.x * 32;
  const int mrow0  = (wave >> 2) * 16;
  const int ntbase = (wave & 3) * 2;

  // Load W_hh into LDS once (reused all 32 steps; +8-half pad keeps the
  // per-WMMA B-fragment b128 reads bank-conflict free, so done manually).
  for (int q = tid; q < 512 * 16; q += 256) {
    int r = q >> 4, k8 = (q & 15) << 3;
    *(ush8*)(Whh + r * KP2 + k8) = *(const ush8*)(whh_bf + (size_t)r * H_ + k8);
  }
  {
    ush8 z = {0, 0, 0, 0, 0, 0, 0, 0};
    for (int q = tid; q < (32 * KP2) / 8; q += 256) *(ush8*)(hbuf + q * 8) = z;
  }
  float creg[2][8];
#pragma unroll
  for (int j = 0; j < 2; ++j)
#pragma unroll
    for (int v = 0; v < 8; ++v) creg[j][v] = 0.f;
  __syncthreads();

  const int mloc = mrow0 + ((lane >> 4) << 3);

  for (int t = 0; t < T_; ++t) {
    // --- Tensor Data Mover: xp[t][b0..b0+32)[0..512) (bf16) -> LDS xpb ----
    // D# group0: count=1 | lds_addr | global_addr(57b) | type=2
    // D# group1: data_size=2B; tensor_dim0=512, tensor_dim1=32;
    //            tile_dim0=512, tile_dim1=32; tensor_dim0_stride=512
    // Tile width == row stride -> one contiguous 32KB burst.
    if (wave == 0) {
      unsigned long long ga =
          (unsigned long long)(const void*)(xp + ((size_t)t * B_ + b0) * G4_);
      u32x4 g0;
      g0[0] = 1u;                                       // count=1 (valid, user)
      g0[1] = XPB_LDS_OFF;                              // lds_addr (bytes)
      g0[2] = (unsigned)(ga & 0xFFFFFFFFu);             // global_addr[31:0]
      g0[3] = (unsigned)((ga >> 32) & 0x01FFFFFFu)      // global_addr[56:32]
              | (2u << 30);                             // type=2 ("image")
      i32x8 g1;
      g1[0] = 0x10000;                                  // data_size=1 -> 2 bytes
      g1[1] = (int)(512u << 16);                        // tensor_dim0 = 512
      g1[2] = (int)(32u << 16);                         // tensor_dim1 = 32
      g1[3] = (int)(512u << 16);                        // tile_dim0 = 512
      g1[4] = 32;                                       // tile_dim1 = 32
      g1[5] = 512;                                      // tensor_dim0_stride
      g1[6] = 0;
      g1[7] = 0;
      i32x4 gz4 = {0, 0, 0, 0};                         // 2D: groups 2/3 unused
      i32x8 gz8 = {0, 0, 0, 0, 0, 0, 0, 0};             // extra group (6-arg form)
      __builtin_amdgcn_tensor_load_to_lds(g0, g1, gz4, gz4, gz8, 0);
      __builtin_amdgcn_s_wait_tensorcnt(0);             // s_wait_tensorcnt 0
    }
    __syncthreads();                                    // publish tile to block

    // h(t-1) as A fragments (wave reads its own 16 rows, all 128 K).
    v16bf ah[4];
#pragma unroll
    for (int kb = 0; kb < 4; ++kb) ah[kb] = ldA(hbuf, KP2, mrow0, kb * 32, lane);
    __syncthreads();   // all ah reads done before anyone rewrites hbuf

#pragma unroll
    for (int j = 0; j < 2; ++j) {
      const int nt = ntbase + j;
      v8f g4[4];
#pragma unroll
      for (int gg = 0; gg < 4; ++gg) {             // gate order i,f,g,o
        const int colb = gg * 128 + nt * 16;
        const int nn   = colb + (lane & 15);
        v8f acc;
#pragma unroll
        for (int v = 0; v < 8; ++v) acc[v] = bf2f(xpb[(mloc + v) * XPP2 + nn]);
#pragma unroll
        for (int kb = 0; kb < 4; ++kb)
          acc = wmma_bf16(ah[kb], ldB(Whh, KP2, colb, kb * 32, lane), acc);
        g4[gg] = acc;
      }
      const int hcol = nt * 16 + (lane & 15);
#pragma unroll
      for (int v = 0; v < 8; ++v) {
        float iv = sigmoidf_(g4[0][v]);
        float fv = sigmoidf_(g4[1][v]);
        float gv = tanhf(g4[2][v]);
        float ov = sigmoidf_(g4[3][v]);
        float cn = fv * creg[j][v] + iv * gv;
        creg[j][v] = cn;
        hbuf[(mloc + v) * KP2 + hcol] = f2bf(ov * tanhf(cn));
      }
    }
    __syncthreads();   // readers done before next step's TDM overwrites xpb
  }

  // Collapsed head: out[m] = h[m,:].v + s
  if (tid < 32) {
    float acc = sbias[0];
    for (int k = 0; k < H_; ++k) acc += bf2f(hbuf[tid * KP2 + k]) * vh[k];
    out[b0 + tid] = acc;
  }
}

// ---------------------------------------------------------------------------
extern "C" void kernel_launch(void* const* d_in, const int* in_sizes, int n_in,
                              void* d_out, int out_size, void* d_ws, size_t ws_size,
                              hipStream_t stream) {
  (void)in_sizes; (void)n_in; (void)out_size; (void)ws_size;
  const float* x   = (const float*)d_in[0];
  const float* Wih = (const float*)d_in[1];
  const float* Whh = (const float*)d_in[2];
  const float* bih = (const float*)d_in[3];
  const float* bhh = (const float*)d_in[4];
  const float* W1  = (const float*)d_in[5];
  const float* b1  = (const float*)d_in[6];
  const float* W2  = (const float*)d_in[7];
  const float* b2  = (const float*)d_in[8];
  float* out = (float*)d_out;

  char* ws = (char*)d_ws;                           // needs ~135 MB scratch
  unsigned short* wih_bf = (unsigned short*)(ws + 0);        // 294,912 B
  unsigned short* whh_bf = (unsigned short*)(ws + 294912);   // 131,072 B
  float* bias  = (float*)(ws + 425984);                      //   2,048 B
  float* vh    = (float*)(ws + 428032);                      //     512 B
  float* sb    = (float*)(ws + 428544);                      //     512 B
  unsigned short* xp = (unsigned short*)(ws + 429056);       // 134,217,728 B

  lstm_prep_kernel<<<64, 256, 0, stream>>>(Wih, Whh, bih, bhh, W1, b1, W2, b2,
                                           wih_bf, whh_bf, bias, vh, sb);

  const size_t lds1 = (size_t)2 * (128 * XPCH + 128 * XPCH);           // 151,552 B
  lstm_xproj_kernel<<<(B_ * T_) / 128, 256, lds1, stream>>>(x, wih_bf, bias, xp);

  const size_t lds2 = (size_t)2 * (512 * KP2 + 32 * KP2 + 32 * XPP2);  // 180,736 B
  lstm_scan_kernel<<<B_ / 32, 256, lds2, stream>>>(xp, whh_bf, vh, sb, out);
}
